// MeshGraphNet_unc_89713276879385
// MI455X (gfx1250) — matrix operand order repaired and checked by reading, not compiled, not run on hardware
//
#include <hip/hip_runtime.h>
#include <hip/hip_bf16.h>

typedef _Float16 half_t;
typedef __attribute__((ext_vector_type(16))) _Float16 v16h;
typedef __attribute__((ext_vector_type(8)))  _Float16 v8h;
typedef __attribute__((ext_vector_type(8)))  float    v8f;
typedef __attribute__((ext_vector_type(4)))  float    v4f;

union V16U { v16h v; v8h h[2]; };

#define D_DIM 128
#define LN_EPS 1e-5f

// ---------------------------------------------------------------------------
// Args for the generic fused 3-layer MLP kernel
// ---------------------------------------------------------------------------
struct MlpArgs {
  const float* in0;      // mode-specific
  const float* in1;
  const float* in2;
  const int*   edges;
  const half_t* w0; const float* b0;   // packed f16 weights (WMMA B layout)
  const half_t* w1; const float* b1;
  const half_t* w2; const float* b2;
  const float* ln_g; const float* ln_b;
  const float* w2f; const float* b2f;  // decoder final 128->2 layer (f32)
  const float* cnt;
  float* agg;
  float* out;
  int M;
};

// ---------------------------------------------------------------------------
// Weight packing: f32 [K x 128] row-major -> f16 in this file's B-operand VGPR
// layout for V_WMMA_F32_16X16X32_F16 (contiguous k per lane half, not ISA table order):
//   lane n (n<16): col = nt*16+n, halves 0..15 = K kc*32+0..15
//   lane n>=16   : col = nt*16+(n-16), halves = K kc*32+16..31
// packed index = (((kc*8 + nt)*32 + lane)*16 + h)  -> one 32B load per lane.
// Zero-pads K to a multiple of 32.
// ---------------------------------------------------------------------------
__global__ void pack_w(const float* __restrict__ src, half_t* __restrict__ dst,
                       int Kreal, int total) {
  int idx = blockIdx.x * 256 + threadIdx.x;
  if (idx >= total) return;
  int h    = idx & 15;
  int lane = (idx >> 4) & 31;
  int nt   = (idx >> 9) & 7;
  int kc   = idx >> 12;
  int n = nt * 16 + (lane & 15);
  int k = kc * 32 + (lane >> 4) * 16 + h;
  float v = (k < Kreal) ? src[k * 128 + n] : 0.f;
  dst[idx] = (half_t)v;
}

__global__ void zero_f32(float* p, int n) {
  int i = blockIdx.x * blockDim.x + threadIdx.x;
  if (i < n) p[i] = 0.f;
}

__global__ void count_edges(const int* __restrict__ edges, float* cnt, int E) {
  int e = blockIdx.x * blockDim.x + threadIdx.x;
  if (e < E) atomicAdd(&cnt[edges[e * 2 + 1]], 1.0f);
}

// ---------------------------------------------------------------------------
// One 16-row x 128-col GEMM step for a wave:  acc += X(16 x 32*kchunks) @ W
// X is in LDS (f16, row-major, ldx halves per row). A fragment matches packed B
// k order: lanes 0-15 rows 0-15 K {0..15}; lanes 16-31 K {16..31}.
// ---------------------------------------------------------------------------
__device__ __forceinline__ void gemm16(const half_t* xw, int ldx,
                                       const half_t* __restrict__ wp,
                                       int kchunks, v8f* acc) {
  const int lane = threadIdx.x & 31;
  const int hi = lane >> 4;
  const int mr = lane & 15;
#pragma unroll
  for (int kc = 0; kc < kchunks; ++kc) {
    V16U a;
    const half_t* ap = xw + mr * ldx + kc * 32 + hi * 16;
    a.h[0] = *(const v8h*)(ap);
    a.h[1] = *(const v8h*)(ap + 8);
    const half_t* bp = wp + (size_t)(kc * 8) * 512 + (size_t)lane * 16;
#pragma unroll
    for (int nt = 0; nt < 8; ++nt) {
      v16h b = *(const v16h*)(bp + nt * 512);
      acc[nt] = __builtin_amdgcn_wmma_f32_16x16x32_f16(
          false, a.v, false, b, (short)0, acc[nt], false, false);
    }
  }
}

__device__ __forceinline__ void zero_acc(v8f* acc) {
#pragma unroll
  for (int nt = 0; nt < 8; ++nt)
#pragma unroll
    for (int q = 0; q < 8; ++q) acc[nt][q] = 0.f;
}

// bias + ReLU, store C-fragment back to per-wave LDS hidden tile (f16)
__device__ __forceinline__ void relu_store_h(v8f* acc, const float* __restrict__ bias,
                                             half_t* hw) {
  const int lane = threadIdx.x & 31;
  const int hi = lane >> 4;
  const int mr = lane & 15;
#pragma unroll
  for (int nt = 0; nt < 8; ++nt) {
    float bv = bias[nt * 16 + mr];
#pragma unroll
    for (int i = 0; i < 8; ++i) {
      float x = acc[nt][i] + bv;
      x = fmaxf(x, 0.f);
      hw[(i + 8 * hi) * D_DIM + nt * 16 + mr] = (half_t)x;
    }
  }
}

// ---------------------------------------------------------------------------
// Generic fused 3-layer MLP. MODE: 0=node enc, 1=edge enc, 2=gnn edge,
// 3=gnn node, 4=decoder. K0C = K-chunks (32 each) of layer-0 input.
// ---------------------------------------------------------------------------
template <int MODE, int K0C, bool LN, bool FINAL2, bool SIG>
__global__ __launch_bounds__(128) void mlp_kernel(MlpArgs a) {
  __shared__ __align__(16) half_t Xs[64 * K0C * 32];
  __shared__ __align__(16) half_t Hs[64 * D_DIM];
  const int base = blockIdx.x * 64;
  const int tid = threadIdx.x;
  const int wave = tid >> 5;
  const int lane = tid & 31;
  const int hi = lane >> 4;
  const int mr = lane & 15;
  const int M = a.M;

  // ---------------- stage layer-0 input tile into LDS (f16) ----------------
  if constexpr (MODE == 0) {  // node encoder: [cond(3), type(2), pos(2)] pad 32
    for (int j = tid; j < 64 * 32; j += 128) {
      int r = j >> 5, c = j & 31;
      int g = base + r; if (g > M - 1) g = M - 1;
      float x = 0.f;
      if (c < 3)      x = a.in0[g * 3 + c];
      else if (c < 5) x = a.in1[g * 2 + (c - 3)];
      else if (c < 7) x = a.in2[g * 2 + (c - 5)];
      Xs[j] = (half_t)x;
    }
  } else if constexpr (MODE == 1) {  // edge encoder: [dx, dy, |d|] pad 32
    for (int j = tid; j < 64 * 32; j += 128) {
      int r = j >> 5, c = j & 31;
      int g = base + r; if (g > M - 1) g = M - 1;
      float x = 0.f;
      if (c < 3) {
        int s = a.edges[g * 2], d = a.edges[g * 2 + 1];
        float dx = a.in2[d * 2]     - a.in2[s * 2];
        float dy = a.in2[d * 2 + 1] - a.in2[s * 2 + 1];
        x = (c == 0) ? dx : (c == 1) ? dy : sqrtf(dx * dx + dy * dy);
      }
      Xs[j] = (half_t)x;
    }
  } else if constexpr (MODE == 2) {  // gnn edge: [V[snd], V[rcv], E]
    for (int j4 = tid; j4 < 64 * 96; j4 += 128) {
      int r = j4 / 96, c4 = (j4 % 96) * 4;
      int g = base + r; if (g > M - 1) g = M - 1;
      v4f f;
      if (c4 < 128) {        // V[snd] gather: reused data -> keep cached
        f = *(const v4f*)(a.in0 + (size_t)a.edges[g * 2] * D_DIM + c4);
      } else if (c4 < 256) { // V[rcv] gather: reused data -> keep cached
        f = *(const v4f*)(a.in0 + (size_t)a.edges[g * 2 + 1] * D_DIM + (c4 - 128));
      } else {               // E row: streamed once -> non-temporal
        f = __builtin_nontemporal_load(
            (const v4f*)(a.in1 + (size_t)g * D_DIM + (c4 - 256)));
      }
      half_t* dst = &Xs[r * 384 + c4];
      dst[0] = (half_t)f.x; dst[1] = (half_t)f.y;
      dst[2] = (half_t)f.z; dst[3] = (half_t)f.w;
    }
  } else if constexpr (MODE == 3) {  // gnn node: [V, agg/max(cnt,1)]
    for (int j4 = tid; j4 < 64 * 64; j4 += 128) {
      int r = j4 >> 6, c4 = (j4 & 63) * 4;
      int g = base + r; if (g > M - 1) g = M - 1;
      v4f f;
      if (c4 < 128) {
        f = *(const v4f*)(a.in0 + (size_t)g * D_DIM + c4);
      } else {
        f = *(const v4f*)(a.in1 + (size_t)g * D_DIM + (c4 - 128));
        float inv = 1.f / fmaxf(a.cnt[g], 1.f);
        f.x *= inv; f.y *= inv; f.z *= inv; f.w *= inv;
      }
      half_t* dst = &Xs[r * 256 + c4];
      dst[0] = (half_t)f.x; dst[1] = (half_t)f.y;
      dst[2] = (half_t)f.z; dst[3] = (half_t)f.w;
    }
  } else {  // decoder: V rows
    for (int j4 = tid; j4 < 64 * 32; j4 += 128) {
      int r = j4 >> 5, c4 = (j4 & 31) * 4;
      int g = base + r; if (g > M - 1) g = M - 1;
      v4f f = *(const v4f*)(a.in0 + (size_t)g * D_DIM + c4);
      half_t* dst = &Xs[r * 128 + c4];
      dst[0] = (half_t)f.x; dst[1] = (half_t)f.y;
      dst[2] = (half_t)f.z; dst[3] = (half_t)f.w;
    }
  }
  __syncthreads();

  half_t* xw = Xs + wave * 16 * (K0C * 32);
  half_t* hw = Hs + wave * 16 * D_DIM;

  v8f acc[8];
  // layer 0 (+ReLU)
  zero_acc(acc);
  gemm16(xw, K0C * 32, a.w0, K0C, acc);
  relu_store_h(acc, a.b0, hw);
  // layer 1 (+ReLU) — in-place on per-wave hidden tile (reads precede writes)
  zero_acc(acc);
  gemm16(hw, D_DIM, a.w1, 4, acc);
  relu_store_h(acc, a.b1, hw);

  if constexpr (FINAL2) {
    // decoder: tiny 128->2 final layer, per-thread from LDS
    __shared__ float W2s[256];
    for (int j = tid; j < 256; j += 128) W2s[j] = a.w2f[j];
    __syncthreads();
    int r = tid >> 1, oc = tid & 1;
    int g = base + r;
    float s = a.b2f[oc];
    const half_t* hr = Hs + r * D_DIM;
#pragma unroll 8
    for (int k = 0; k < D_DIM; ++k) s += (float)hr[k] * W2s[k * 2 + oc];
    if (SIG) s = 1.f / (1.f + expf(-s));
    if (g < M) a.out[(size_t)g * 2 + oc] = s;
    return;
  }

  // layer 2 (no ReLU)
  zero_acc(acc);
  gemm16(hw, D_DIM, a.w2, 4, acc);
#pragma unroll
  for (int nt = 0; nt < 8; ++nt) {
    float bv = a.b2[nt * 16 + mr];
#pragma unroll
    for (int i = 0; i < 8; ++i) acc[nt][i] += bv;
  }

  if constexpr (LN) {
    float gv[8], bev[8];
#pragma unroll
    for (int nt = 0; nt < 8; ++nt) {
      gv[nt] = a.ln_g[nt * 16 + mr];
      bev[nt] = a.ln_b[nt * 16 + mr];
    }
#pragma unroll
    for (int i = 0; i < 8; ++i) {
      float s1 = 0.f, s2 = 0.f;
#pragma unroll
      for (int nt = 0; nt < 8; ++nt) { float x = acc[nt][i]; s1 += x; s2 += x * x; }
      // reduce across the 16 lanes holding this row (xor 1,2,4,8 stays in half)
#pragma unroll
      for (int m = 1; m < 16; m <<= 1) {
        s1 += __shfl_xor(s1, m, 32);
        s2 += __shfl_xor(s2, m, 32);
      }
      float mean = s1 * (1.f / 128.f);
      float var = s2 * (1.f / 128.f) - mean * mean;
      float rstd = rsqrtf(var + LN_EPS);
#pragma unroll
      for (int nt = 0; nt < 8; ++nt)
        acc[nt][i] = (acc[nt][i] - mean) * rstd * gv[nt] + bev[nt];
    }
  }

  // ---------------- epilogue store ----------------
#pragma unroll
  for (int i = 0; i < 8; ++i) {
    int lr = wave * 16 + i + 8 * hi;
    int g = base + lr;
    if (g < M) {
      if constexpr (MODE == 2) {
        int cn = a.edges[g * 2 + 1];
#pragma unroll
        for (int nt = 0; nt < 8; ++nt) {
          int c = nt * 16 + mr;
          float y = acc[nt][i];
          // E += e_emb : E is streamed -> non-temporal read + write-rinse store
          float eo = __builtin_nontemporal_load(&a.in1[(size_t)g * D_DIM + c]);
          __builtin_nontemporal_store(eo + y, &a.out[(size_t)g * D_DIM + c]);
          atomicAdd(&a.agg[(size_t)cn * D_DIM + c], y);  // scatter-sum (L2 atomic)
        }
      } else if constexpr (MODE == 3) {
#pragma unroll
        for (int nt = 0; nt < 8; ++nt) {
          int c = nt * 16 + mr;
          a.out[(size_t)g * D_DIM + c] = a.in0[(size_t)g * D_DIM + c] + acc[nt][i];  // V += v_emb
        }
      } else if constexpr (MODE == 1) {
#pragma unroll
        for (int nt = 0; nt < 8; ++nt)  // E encoder output: streamed -> NT store
          __builtin_nontemporal_store(acc[nt][i],
                                      &a.out[(size_t)g * D_DIM + nt * 16 + mr]);
      } else {
#pragma unroll
        for (int nt = 0; nt < 8; ++nt)
          a.out[(size_t)g * D_DIM + nt * 16 + mr] = acc[nt][i];
      }
    }
  }
}

// ---------------------------------------------------------------------------
// Host-side orchestration
// ---------------------------------------------------------------------------
extern "C" void kernel_launch(void* const* d_in, const int* in_sizes, int n_in,
                              void* d_out, int out_size, void* d_ws, size_t ws_size,
                              hipStream_t stream) {
  // Detect flatten order: insertion order (node_pos first, 100000 elems) vs
  // jax tree-sorted (conditions_input first, 150000 elems).
  int i_pos, i_type, i_cond, i_edges;
  int FV, FE, GN, DP, DU;
  const int P = 4;
  if (in_sizes[0] == 150000) {  // sorted keys
    i_cond = 0; i_edges = 1; i_pos = 2; i_type = 3;
    DP = P; DU = P + 6; FE = P + 12; FV = P + 20; GN = P + 28;
  } else {                       // insertion order
    i_pos = 0; i_type = 1; i_cond = 2; i_edges = 3;
    FV = P; FE = P + 8; GN = P + 16; DP = P + 208; DU = P + 214;
  }
  const int Nn = in_sizes[i_pos] / 2;
  const int Ee = in_sizes[i_edges] / 2;

  auto f = [&](int i) { return (const float*)d_in[i]; };
  const float* node_pos  = f(i_pos);
  const float* node_type = f(i_type);
  const float* cond      = f(i_cond);
  const int*   edges     = (const int*)d_in[i_edges];

  // ---- pack all GEMM weights to f16 WMMA-B layout in workspace ----
  struct PI { int idx; int K; };
  PI items[82];
  int ni = 0;
  items[ni++] = {FV + 0, 7};   items[ni++] = {FV + 2, 128}; items[ni++] = {FV + 4, 128};
  items[ni++] = {FE + 0, 3};   items[ni++] = {FE + 2, 128}; items[ni++] = {FE + 4, 128};
  for (int l = 0; l < 12; ++l) {
    int b = GN + l * 16;
    items[ni++] = {b + 0, 384}; items[ni++] = {b + 2, 128}; items[ni++] = {b + 4, 128};
    items[ni++] = {b + 8, 256}; items[ni++] = {b + 10, 128}; items[ni++] = {b + 12, 128};
  }
  items[ni++] = {DP + 0, 128}; items[ni++] = {DP + 2, 128};
  items[ni++] = {DU + 0, 128}; items[ni++] = {DU + 2, 128};  // ni == 82

  char* ws = (char*)d_ws;
  half_t* wbase = (half_t*)ws;
  size_t woff[82];
  size_t cur = 0;
  for (int i = 0; i < ni; ++i) {
    int chunks = (items[i].K + 31) / 32;
    size_t total = (size_t)chunks * 4096;  // halves
    woff[i] = cur;
    cur += total;
    int blocks = (int)((total + 255) / 256);
    pack_w<<<blocks, 256, 0, stream>>>(f(items[i].idx), wbase + woff[i],
                                       items[i].K, (int)total);
  }
  size_t o = (cur * sizeof(half_t) + 255) & ~(size_t)255;
  float* V   = (float*)(ws + o); o += (size_t)Nn * D_DIM * 4;
  float* E   = (float*)(ws + o); o += (size_t)Ee * D_DIM * 4;
  float* AG  = (float*)(ws + o); o += (size_t)Nn * D_DIM * 4;
  float* CNT = (float*)(ws + o); o += (size_t)Nn * 4;

  // ---- degree counts (loop-invariant) ----
  zero_f32<<<(Nn + 255) / 256, 256, 0, stream>>>(CNT, Nn);
  count_edges<<<(Ee + 255) / 256, 256, 0, stream>>>(edges, CNT, Ee);

  const int nblk = (Nn + 63) / 64;
  const int eblk = (Ee + 63) / 64;

  // ---- encoders ----
  {
    MlpArgs an = {};
    an.in0 = cond; an.in1 = node_type; an.in2 = node_pos;
    an.w0 = wbase + woff[0]; an.b0 = f(FV + 1);
    an.w1 = wbase + woff[1]; an.b1 = f(FV + 3);
    an.w2 = wbase + woff[2]; an.b2 = f(FV + 5);
    an.ln_g = f(FV + 6); an.ln_b = f(FV + 7);
    an.out = V; an.M = Nn;
    mlp_kernel<0, 1, true, false, false><<<nblk, 128, 0, stream>>>(an);
  }
  {
    MlpArgs ae = {};
    ae.in2 = node_pos; ae.edges = edges;
    ae.w0 = wbase + woff[3]; ae.b0 = f(FE + 1);
    ae.w1 = wbase + woff[4]; ae.b1 = f(FE + 3);
    ae.w2 = wbase + woff[5]; ae.b2 = f(FE + 5);
    ae.ln_g = f(FE + 6); ae.ln_b = f(FE + 7);
    ae.out = E; ae.M = Ee;
    mlp_kernel<1, 1, true, false, false><<<eblk, 128, 0, stream>>>(ae);
  }

  // ---- processor: 12 message-passing layers ----
  int widx = 6;
  for (int l = 0; l < 12; ++l) {
    int b = GN + l * 16;
    zero_f32<<<(Nn * D_DIM + 255) / 256, 256, 0, stream>>>(AG, Nn * D_DIM);
    {
      MlpArgs aE = {};
      aE.in0 = V; aE.in1 = E; aE.edges = edges;
      aE.w0 = wbase + woff[widx + 0]; aE.b0 = f(b + 1);
      aE.w1 = wbase + woff[widx + 1]; aE.b1 = f(b + 3);
      aE.w2 = wbase + woff[widx + 2]; aE.b2 = f(b + 5);
      aE.ln_g = f(b + 6); aE.ln_b = f(b + 7);
      aE.agg = AG; aE.out = E; aE.M = Ee;
      mlp_kernel<2, 12, true, false, false><<<eblk, 128, 0, stream>>>(aE);
    }
    {
      MlpArgs aN = {};
      aN.in0 = V; aN.in1 = AG; aN.cnt = CNT;
      aN.w0 = wbase + woff[widx + 3]; aN.b0 = f(b + 9);
      aN.w1 = wbase + woff[widx + 4]; aN.b1 = f(b + 11);
      aN.w2 = wbase + woff[widx + 5]; aN.b2 = f(b + 13);
      aN.ln_g = f(b + 14); aN.ln_b = f(b + 15);
      aN.out = V; aN.M = Nn;
      mlp_kernel<3, 8, true, false, false><<<nblk, 128, 0, stream>>>(aN);
    }
    widx += 6;
  }

  // ---- decoders ----
  {
    MlpArgs ap = {};
    ap.in0 = V;
    ap.w0 = wbase + woff[78]; ap.b0 = f(DP + 1);
    ap.w1 = wbase + woff[79]; ap.b1 = f(DP + 3);
    ap.w2f = f(DP + 4); ap.b2f = f(DP + 5);
    ap.out = (float*)d_out; ap.M = Nn;
    mlp_kernel<4, 4, false, true, false><<<nblk, 128, 0, stream>>>(ap);
  }
  {
    MlpArgs au = {};
    au.in0 = V;
    au.w0 = wbase + woff[80]; au.b0 = f(DU + 1);
    au.w1 = wbase + woff[81]; au.b1 = f(DU + 3);
    au.w2f = f(DU + 4); au.b2f = f(DU + 5);
    au.out = (float*)d_out + (size_t)Nn * 2; au.M = Nn;
    mlp_kernel<4, 4, false, true, true><<<nblk, 128, 0, stream>>>(au);
  }
}
